// PointNet2_Segm_88828513616119
// MI455X (gfx1250) — matrix operand order repaired and checked by reading, not compiled, not run on hardware
//
#include <hip/hip_runtime.h>
#include <hip/hip_bf16.h>
#include <cstdint>
#include <cstddef>

#define BATCH 16
#define NRAW 1300
#define NPTS 1162
#define NS1 581
#define NS2 146
#define KNB 64

typedef _Float16 h16;
typedef __attribute__((ext_vector_type(16))) _Float16 v16h;
typedef __attribute__((ext_vector_type(8)))  _Float16 v8h;
typedef __attribute__((ext_vector_type(8)))  float    v8f;
typedef int v2i __attribute__((__vector_size__(8)));

#define AS1 __attribute__((address_space(1)))
#define AS3 __attribute__((address_space(3)))

#ifndef __has_builtin
#define __has_builtin(x) 0
#endif

#if __has_builtin(__builtin_amdgcn_global_load_async_to_lds_b64)
#define USE_ASYNC_LDS 1
#else
#define USE_ASYNC_LDS 0
#endif

__device__ __forceinline__ void wait_async0() {
#if __has_builtin(__builtin_amdgcn_s_wait_asynccnt)
  __builtin_amdgcn_s_wait_asynccnt(0);
#else
  asm volatile("s_wait_asynccnt 0x0" ::: "memory");
#endif
}

// ---------------------------------------------------------------------------
// prep: pos0 (B,NPTS,3) f32 ; feat0 (B,NPTS,3) f16
// ---------------------------------------------------------------------------
__global__ void k_prep(const float* __restrict__ pc, float* __restrict__ pos0,
                       h16* __restrict__ feat0) {
  int t = blockIdx.x * blockDim.x + threadIdx.x;
  if (t >= BATCH * NPTS) return;
  int b = t / NPTS, i = t % NPTS;
  const float* src = pc + ((size_t)b * NRAW + i) * 6;
  float* p = pos0 + (size_t)t * 3;
  p[0] = src[0]; p[1] = src[1]; p[2] = src[2];
  h16* f = feat0 + (size_t)t * 3;
  if (i < 1024) {
    f[0] = (h16)0.f; f[1] = (h16)0.f; f[2] = (h16)0.f;
  } else {
    int j = i - 1024;
    const float* a = pc + ((size_t)b * NRAW + NPTS + j) * 6;
    f[0] = (h16)(a[0] - src[0]);
    f[1] = (h16)(a[1] - src[1]);
    f[2] = (h16)(a[2] - src[2]);
  }
}

// ---------------------------------------------------------------------------
// farthest point sampling: one block per batch, LDS min-distance + argmax
// ---------------------------------------------------------------------------
__global__ void k_fps(const float* __restrict__ pos, int N, int nsamp,
                      int* __restrict__ sel) {
  __shared__ float mind[NRAW];
  __shared__ float rmax[256];
  __shared__ int   rarg[256];
  __shared__ float lastp[3];
  int b = blockIdx.x, tid = threadIdx.x, T = blockDim.x;
  const float* P = pos + (size_t)b * N * 3;
  int* S = sel + (size_t)b * nsamp;
  for (int j = tid; j < N; j += T) mind[j] = 3.0e38f;
  if (tid == 0) { S[0] = 0; lastp[0] = P[0]; lastp[1] = P[1]; lastp[2] = P[2]; }
  __syncthreads();
  for (int i = 1; i < nsamp; i++) {
    float lx = lastp[0], ly = lastp[1], lz = lastp[2];
    float best = -1.f; int barg = 0;
    for (int j = tid; j < N; j += T) {
      float dx = P[j*3+0]-lx, dy = P[j*3+1]-ly, dz = P[j*3+2]-lz;
      float d = dx*dx + dy*dy + dz*dz;
      float m = fminf(mind[j], d); mind[j] = m;
      if (m > best) { best = m; barg = j; }
    }
    rmax[tid] = best; rarg[tid] = barg;
    __syncthreads();
    for (int s = T >> 1; s > 0; s >>= 1) {
      if (tid < s) {
        float ov = rmax[tid+s]; int oi = rarg[tid+s];
        if (ov > rmax[tid] || (ov == rmax[tid] && oi < rarg[tid])) {
          rmax[tid] = ov; rarg[tid] = oi;
        }
      }
      __syncthreads();
    }
    if (tid == 0) {
      int w = rarg[0]; S[i] = w;
      lastp[0] = P[w*3+0]; lastp[1] = P[w*3+1]; lastp[2] = P[w*3+2];
    }
    __syncthreads();
  }
}

__global__ void k_gather_pos(const float* __restrict__ src, const int* __restrict__ idx,
                             float* __restrict__ dst, int Nc, int Nsrc) {
  int t = blockIdx.x * blockDim.x + threadIdx.x;
  if (t >= BATCH * Nc) return;
  int b = t / Nc;
  int j = idx[t];
  const float* s = src + ((size_t)b * Nsrc + j) * 3;
  float* d = dst + (size_t)t * 3;
  d[0] = s[0]; d[1] = s[1]; d[2] = s[2];
}

// ---------------------------------------------------------------------------
// radius-limited 64-NN per center: one block per center, LDS d2 + 64x argmin
// ---------------------------------------------------------------------------
__global__ void k_neighbors(const float* __restrict__ pos, const float* __restrict__ ctr,
                            int Nsrc, int Nc, float r2,
                            int* __restrict__ nidx, int* __restrict__ nval) {
  __shared__ float d2s[NRAW];
  __shared__ float rmin[128];
  __shared__ int   rarg[128];
  int cb = blockIdx.x; int b = cb / Nc;
  int tid = threadIdx.x; const int T = 128;
  const float* P = pos + (size_t)b * Nsrc * 3;
  const float* C = ctr + (size_t)cb * 3;
  float cx = C[0], cy = C[1], cz = C[2];
  for (int j = tid; j < Nsrc; j += T) {
    float dx = P[j*3]-cx, dy = P[j*3+1]-cy, dz = P[j*3+2]-cz;
    d2s[j] = dx*dx + dy*dy + dz*dz;
  }
  __syncthreads();
  int* oI = nidx + (size_t)cb * KNB;
  int* oV = nval + (size_t)cb * KNB;
  for (int k = 0; k < KNB; k++) {
    float best = 3.0e38f; int barg = 0;
    for (int j = tid; j < Nsrc; j += T) {
      float v = d2s[j];
      if (v < best) { best = v; barg = j; }
    }
    rmin[tid] = best; rarg[tid] = barg;
    __syncthreads();
    for (int s = T >> 1; s > 0; s >>= 1) {
      if (tid < s) {
        float ov = rmin[tid+s]; int oi = rarg[tid+s];
        if (ov < rmin[tid] || (ov == rmin[tid] && oi < rarg[tid])) {
          rmin[tid] = ov; rarg[tid] = oi;
        }
      }
      __syncthreads();
    }
    if (tid == 0) {
      if (rmin[0] <= r2) { oI[k] = rarg[0]; oV[k] = 1; d2s[rarg[0]] = 3.0e38f; }
      else               { oI[k] = 0;       oV[k] = 0; }
    }
    __syncthreads();
  }
}

// ---------------------------------------------------------------------------
// weight repack into per-lane WMMA B-fragment layout (contiguous k = 16*(lane/16)+j, not ISA table order):
// packed[((kt*(Np/16)+nt)*32 + lane)*16 + j] = W[kt*32 + (lane>=16?16:0)+j][nt*16+(lane&15)]
// ---------------------------------------------------------------------------
__global__ void k_pack_w(const float* __restrict__ W, int K, int N, int Kp, int Np,
                         h16* __restrict__ Wp) {
  int t = blockIdx.x * blockDim.x + threadIdx.x;
  if (t >= Kp * Np) return;
  int j = t & 15;
  int lane = (t >> 4) & 31;
  int tile = t >> 9;
  int ntn = Np >> 4;
  int nt = tile % ntn, kt = tile / ntn;
  int n = (nt << 4) + (lane & 15);
  int k = (kt << 5) + ((lane >> 4) << 4) + j;
  float v = (k < K && n < N) ? W[(size_t)k * N + n] : 0.f;
  Wp[t] = (h16)v;
}

__global__ void k_pack_b(const float* __restrict__ bsrc, int N, int Np, float* __restrict__ bp) {
  int t = blockIdx.x * blockDim.x + threadIdx.x;
  if (t >= Np) return;
  bp[t] = (t < N) ? bsrc[t] : 0.f;
}

// ---------------------------------------------------------------------------
// WMMA GEMM: block = 4 waves; each wave computes 4 16x16 tiles of one N-tile
// column (64 rows).  M is pre-padded to a multiple of 256 rows by the host, so
// there are NO bounds checks and the K-loop body is branch-free.  All 4 A
// fragments are loaded back-to-back (partial loadcnt waits), then 4 WMMAs
// issue as a burst on independent accumulators.  B fragments are staged
// through LDS with double-buffered async global->LDS copies.  Y = act(X*W+b)
// ---------------------------------------------------------------------------
union AFrag { v16h v; v8h h[2]; };

__global__ void k_wmma_gemm(const h16* __restrict__ X, const h16* __restrict__ Wp,
                            const float* __restrict__ bp, h16* __restrict__ Yh,
                            float* __restrict__ Yf, int Kp, int Np, int relu) {
#if USE_ASYNC_LDS
  __shared__ h16 ldsB[2][512];
#endif
  const int tid  = threadIdx.x;
  const int lane = tid & 31;
  const int w    = tid >> 5;
  const int nt   = blockIdx.y;
  const int lo = lane & 15, hi = lane >> 4;
  const int ntn = Np >> 4;
  const int ksteps = Kp >> 5;

  const int mt0 = blockIdx.x * 16 + w * 4;   // first of 4 M-tiles for this wave
  const h16* A[4];
  v8f acc[4];
#pragma unroll
  for (int i = 0; i < 4; i++) {
    A[i] = X + (size_t)((mt0 + i) * 16 + lo) * Kp + hi * 16;
#pragma unroll
    for (int r = 0; r < 8; r++) acc[i][r] = 0.f;
  }

#if USE_ASYNC_LDS
  {  // preload B k-tile 0 (1 KiB, 8 B per thread) into buffer 0
    h16* gsrc = (h16*)(Wp + ((size_t)nt) * 512 + tid * 4);
    __builtin_amdgcn_global_load_async_to_lds_b64(
        (AS1 v2i*)gsrc, (AS3 v2i*)&ldsB[0][tid * 4], 0, 0);
  }
#endif

#pragma unroll 2
  for (int kt = 0; kt < ksteps; kt++) {
    int ktn = (kt + 1 < ksteps) ? (kt + 1) : kt;   // clamped, branch-free
    v16h bf;
#if USE_ASYNC_LDS
    wait_async0();          // own async portion complete
    __syncthreads();        // whole tile present; prev reads done before overwrite
    {  // unconditionally fetch next (clamped) k-tile into the other buffer
      h16* gsrc = (h16*)(Wp + ((size_t)ktn * ntn + nt) * 512 + tid * 4);
      __builtin_amdgcn_global_load_async_to_lds_b64(
          (AS1 v2i*)gsrc, (AS3 v2i*)&ldsB[(kt + 1) & 1][tid * 4], 0, 0);
    }
    bf = *(const v16h*)&ldsB[kt & 1][lane * 16];
#else
    bf = *(const v16h*)(Wp + (((size_t)kt * ntn + nt) * 32 + lane) * 16);
#endif
    // phase 1: stage all 4 A fragments (8 b128 loads back-to-back)
    AFrag af[4];
#pragma unroll
    for (int i = 0; i < 4; i++) {
      af[i].h[0] = *(const v8h*)(A[i] + kt * 32);
      af[i].h[1] = *(const v8h*)(A[i] + kt * 32 + 8);
      __builtin_prefetch(A[i] + ktn * 32, 0, 3);
    }
    // phase 2: WMMA burst on independent accumulators
#pragma unroll
    for (int i = 0; i < 4; i++) {
      acc[i] = __builtin_amdgcn_wmma_f32_16x16x32_f16(
          /*neg_a=*/false, af[i].v, /*neg_b=*/false, bf,
          /*c_mod=*/(short)0, acc[i], /*reuse_a=*/false, /*reuse_b=*/false);
    }
  }

  float bias = bp[nt * 16 + lo];
#pragma unroll
  for (int i = 0; i < 4; i++) {
    int m0 = (mt0 + i) * 16;
#pragma unroll
    for (int r = 0; r < 8; r++) {
      float v = acc[i][r] + bias;
      if (relu) v = fmaxf(v, 0.f);
      int row = m0 + r + hi * 8;
      size_t o = (size_t)row * Np + nt * 16 + lo;
      if (Yh) Yh[o] = (h16)v;
      else    Yf[o] = v;
    }
  }
}

// ---------------------------------------------------------------------------
// sa_conv input build: rows (c,k) for one batch, cols [xj | pj-center | 0pad]
// ---------------------------------------------------------------------------
__global__ void k_build_sa_in(const h16* __restrict__ xsrc, int Cx,
                              const float* __restrict__ pos, const float* __restrict__ ctr,
                              const int* __restrict__ nidx, int Nc, int Nsrc, int Kp,
                              h16* __restrict__ Xout, int b) {
  int Mb = Nc * KNB;
  int t = blockIdx.x * blockDim.x + threadIdx.x;
  if (t >= Mb * Kp) return;
  int row = t / Kp, colc = t % Kp;
  int ci = row / KNB, k = row % KNB;
  int j = nidx[((size_t)(b * Nc + ci)) * KNB + k];
  h16 v = (h16)0.f;
  if (colc < Cx) {
    v = xsrc[((size_t)b * Nsrc + j) * Cx + colc];
  } else if (colc < Cx + 3) {
    int d = colc - Cx;
    v = (h16)(pos[((size_t)b * Nsrc + j) * 3 + d] - ctr[((size_t)(b * Nc + ci)) * 3 + d]);
  }
  Xout[t] = v;
}

__global__ void k_maxpool_nbr(const h16* __restrict__ Y, const int* __restrict__ nval,
                              int Nc, int C, h16* __restrict__ xout, int b) {
  int t = blockIdx.x * blockDim.x + threadIdx.x;
  if (t >= Nc * C) return;
  int ci = t / C, ch = t % C;
  const int* V = nval + ((size_t)(b * Nc + ci)) * KNB;
  float m = -3.0e38f;
  for (int k = 0; k < KNB; k++) {
    if (V[k]) m = fmaxf(m, (float)Y[((size_t)(ci * KNB + k)) * C + ch]);
  }
  xout[((size_t)(b * Nc + ci)) * C + ch] = (h16)m;
}

__global__ void k_maxpool_rows(const h16* __restrict__ Y, int Npt, int C, h16* __restrict__ out) {
  int t = blockIdx.x * blockDim.x + threadIdx.x;
  if (t >= BATCH * C) return;
  int b = t / C, ch = t % C;
  float m = -3.0e38f;
  for (int i = 0; i < Npt; i++) m = fmaxf(m, (float)Y[((size_t)(b * Npt + i)) * C + ch]);
  out[t] = (h16)m;
}

__global__ void k_build_sa3in(const h16* __restrict__ x2, const float* __restrict__ pos2,
                              h16* __restrict__ Xout) {
  const int Kp = 288;
  int t = blockIdx.x * blockDim.x + threadIdx.x;
  if (t >= BATCH * NS2 * Kp) return;
  int row = t / Kp, colc = t % Kp;
  h16 v = (h16)0.f;
  if (colc < 256) v = x2[(size_t)row * 256 + colc];
  else if (colc < 259) v = (h16)pos2[(size_t)row * 3 + (colc - 256)];
  Xout[t] = v;
}

__global__ void k_build_fp3in(const h16* __restrict__ g, const h16* __restrict__ x2,
                              h16* __restrict__ Xout) {
  const int Kp = 1280;
  int t = blockIdx.x * blockDim.x + threadIdx.x;
  if (t >= BATCH * NS2 * Kp) return;
  int row = t / Kp, colc = t % Kp;
  int b = row / NS2;
  h16 v;
  if (colc < 1024) v = g[(size_t)b * 1024 + colc];
  else             v = x2[(size_t)row * 256 + (colc - 1024)];
  Xout[t] = v;
}

// 3-NN (first-occurrence ties) with inverse-distance weights
__global__ void k_knn3(const float* __restrict__ pdst, const float* __restrict__ psrc,
                       int Nd, int Ns, int* __restrict__ kidx, float* __restrict__ kw) {
  int t = blockIdx.x * blockDim.x + threadIdx.x;
  if (t >= BATCH * Nd) return;
  int b = t / Nd;
  const float* D = pdst + (size_t)t * 3;
  const float* S = psrc + (size_t)b * Ns * 3;
  float d0 = 3e38f, d1 = 3e38f, d2v = 3e38f; int i0 = 0, i1 = 0, i2 = 0;
  float x = D[0], y = D[1], z = D[2];
  for (int j = 0; j < Ns; j++) {
    float dx = S[j*3]-x, dy = S[j*3+1]-y, dz = S[j*3+2]-z;
    float d = dx*dx + dy*dy + dz*dz;
    if (d < d0)       { d2v = d1; i2 = i1; d1 = d0; i1 = i0; d0 = d; i0 = j; }
    else if (d < d1)  { d2v = d1; i2 = i1; d1 = d; i1 = j; }
    else if (d < d2v) { d2v = d; i2 = j; }
  }
  kidx[t*3+0] = i0; kidx[t*3+1] = i1; kidx[t*3+2] = i2;
  kw[t*3+0] = 1.f / fmaxf(d0, 1e-16f);
  kw[t*3+1] = 1.f / fmaxf(d1, 1e-16f);
  kw[t*3+2] = 1.f / fmaxf(d2v, 1e-16f);
}

__global__ void k_build_fp2in(const h16* __restrict__ xs, const int* __restrict__ kidx,
                              const float* __restrict__ kw, const h16* __restrict__ x1,
                              h16* __restrict__ Xout) {
  const int Kp = 384;
  int t = blockIdx.x * blockDim.x + threadIdx.x;
  if (t >= BATCH * NS1 * Kp) return;
  int row = t / Kp, colc = t % Kp;
  int b = row / NS1;
  h16 v;
  if (colc < 256) {
    const int* I = kidx + (size_t)row * 3;
    const float* W = kw + (size_t)row * 3;
    float ws = W[0] + W[1] + W[2];
    float a = W[0] * (float)xs[((size_t)(b * NS2 + I[0])) * 256 + colc]
            + W[1] * (float)xs[((size_t)(b * NS2 + I[1])) * 256 + colc]
            + W[2] * (float)xs[((size_t)(b * NS2 + I[2])) * 256 + colc];
    v = (h16)(a / ws);
  } else {
    v = x1[(size_t)row * 128 + (colc - 256)];
  }
  Xout[t] = v;
}

__global__ void k_build_fp1in(const h16* __restrict__ xs, const int* __restrict__ kidx,
                              const float* __restrict__ kw, const h16* __restrict__ feat0,
                              h16* __restrict__ Xout) {
  const int Kp = 160;
  int t = blockIdx.x * blockDim.x + threadIdx.x;
  if (t >= BATCH * NPTS * Kp) return;
  int row = t / Kp, colc = t % Kp;
  int b = row / NPTS;
  h16 v = (h16)0.f;
  if (colc < 128) {
    const int* I = kidx + (size_t)row * 3;
    const float* W = kw + (size_t)row * 3;
    float ws = W[0] + W[1] + W[2];
    float a = W[0] * (float)xs[((size_t)(b * NS1 + I[0])) * 128 + colc]
            + W[1] * (float)xs[((size_t)(b * NS1 + I[1])) * 128 + colc]
            + W[2] * (float)xs[((size_t)(b * NS1 + I[2])) * 128 + colc];
    v = (h16)(a / ws);
  } else if (colc < 131) {
    v = feat0[(size_t)row * 3 + (colc - 128)];
  }
  Xout[t] = v;
}

__global__ void k_mean(const h16* __restrict__ X, float* __restrict__ out) {
  int t = blockIdx.x * blockDim.x + threadIdx.x;
  if (t >= BATCH * 128) return;
  int b = t / 128, ch = t % 128;
  float s = 0.f;
  for (int i = 0; i < NPTS; i++) s += (float)X[((size_t)(b * NPTS + i)) * 128 + ch];
  out[t] = s / (float)NPTS;
}

__global__ void k_build_statein(const float* __restrict__ ep, const float* __restrict__ eq,
                                const float* __restrict__ gq, h16* __restrict__ Xout) {
  const int Kp = 32;
  int t = blockIdx.x * blockDim.x + threadIdx.x;
  if (t >= BATCH * Kp) return;
  int b = t / Kp, colc = t % Kp;
  float v = 0.f;
  if (colc < 3) v = ep[b * 3 + colc];
  else if (colc < 7) v = eq[b * 4 + colc - 3];
  else if (colc < 9) v = gq[b * 2 + colc - 7];
  Xout[t] = (h16)v;
}

__global__ void k_build_fingerin(const float* __restrict__ mean, const h16* __restrict__ sfeat,
                                 h16* __restrict__ Xout) {
  const int Kp = 192;
  int t = blockIdx.x * blockDim.x + threadIdx.x;
  if (t >= BATCH * Kp) return;
  int b = t / Kp, colc = t % Kp;
  h16 v;
  if (colc < 128) v = (h16)mean[b * 128 + colc];
  else            v = sfeat[b * 64 + colc - 128];
  Xout[t] = v;
}

__global__ void k_extract_flow(const float* __restrict__ Y, float* __restrict__ out) {
  int t = blockIdx.x * blockDim.x + threadIdx.x;
  if (t >= BATCH * NPTS * 3) return;
  int row = t / 3, c = t % 3;
  out[t] = Y[(size_t)row * 16 + c];
}

__global__ void k_extract_finger(const float* __restrict__ Y, float* __restrict__ out) {
  int b = threadIdx.x;
  if (b >= BATCH) return;
  out[b] = Y[b * 16];
}

// ---------------------------------------------------------------------------
// host driver
// ---------------------------------------------------------------------------
extern "C" void kernel_launch(void* const* d_in, const int* in_sizes, int n_in,
                              void* d_out, int out_size, void* d_ws, size_t ws_size,
                              hipStream_t stream) {
  (void)in_sizes; (void)n_in; (void)out_size; (void)ws_size;
  const float* pc   = (const float*)d_in[0];
  const float* eefp = (const float*)d_in[1];
  const float* eefq = (const float*)d_in[2];
  const float* grip = (const float*)d_in[3];

  struct L { const float* W; const float* b; int K, N, Kp, Np; h16* Wp; float* bp; };
  L Ls[22];
  int li = 0, di = 4;
  auto f32in = [&](int i) { return (const float*)d_in[i]; };
  auto addchain = [&](const int* dims, int nl) {
    for (int i = 0; i < nl; i++) {
      int K = dims[i], N = dims[i + 1];
      int Kp = (K + 31) & ~31, Np = (N + 15) & ~15;
      Ls[li++] = { f32in(di), f32in(di + 1), K, N, Kp, Np, nullptr, nullptr };
      di += 2;
    }
  };
  // params flattened: dict keys sorted (finger, fp1, fp2, fp3, head, sa1, sa2, sa3, state)
  const int d_finger[2] = {192, 1};             addchain(d_finger, 1);  // L0
  const int d_fp1[4]    = {131, 128, 128, 128}; addchain(d_fp1, 3);     // L1..3
  const int d_fp2[3]    = {384, 256, 128};      addchain(d_fp2, 2);     // L4..5
  const int d_fp3[3]    = {1280, 256, 256};     addchain(d_fp3, 2);     // L6..7
  const int d_head[4]   = {128, 128, 128, 3};   addchain(d_head, 3);    // L8..10
  const int d_sa1[4]    = {6, 64, 64, 128};     addchain(d_sa1, 3);     // L11..13
  const int d_sa2[4]    = {131, 128, 128, 256}; addchain(d_sa2, 3);     // L14..16
  const int d_sa3[4]    = {259, 256, 512, 1024};addchain(d_sa3, 3);     // L17..19
  const int d_state[3]  = {9, 64, 64};          addchain(d_state, 2);   // L20..21

  char* base = (char*)d_ws; size_t off = 0;
  auto alloc = [&](size_t bytes) -> void* {
    void* p = base + off;
    off = (off + bytes + 255) & ~(size_t)255;
    return p;
  };
  // padded row counts (multiples of 256) for buffers touched directly by GEMMs
  const int MP_SA3  = 2560;   // 2336 -> 2560  (also fp3)
  const int MP_FP2  = 9472;   // 9296 -> 9472
  const int MP_FP1  = 18688;  // 18592 -> 18688 (also head)
  const int MP_TINY = 256;    // 16 -> 256 (state/finger)

  float* pos0 = (float*)alloc((size_t)BATCH * NPTS * 3 * 4);
  h16*  feat0 = (h16*)alloc((size_t)BATCH * NPTS * 3 * 2);
  int*   idx1 = (int*)alloc((size_t)BATCH * NS1 * 4);
  float* pos1 = (float*)alloc((size_t)BATCH * NS1 * 3 * 4);
  int*   idx2 = (int*)alloc((size_t)BATCH * NS2 * 4);
  float* pos2 = (float*)alloc((size_t)BATCH * NS2 * 3 * 4);
  int*   n1i  = (int*)alloc((size_t)BATCH * NS1 * KNB * 4);
  int*   n1v  = (int*)alloc((size_t)BATCH * NS1 * KNB * 4);
  int*   n2i  = (int*)alloc((size_t)BATCH * NS2 * KNB * 4);
  int*   n2v  = (int*)alloc((size_t)BATCH * NS2 * KNB * 4);
  h16*   x1   = (h16*)alloc((size_t)BATCH * NS1 * 128 * 2);
  h16*   x2   = (h16*)alloc((size_t)BATCH * NS2 * 256 * 2);
  h16*   g    = (h16*)alloc((size_t)BATCH * 1024 * 2);
  h16*   xfp3 = (h16*)alloc((size_t)MP_SA3 * 256 * 2);   // GEMM output: padded
  h16*   xfp2 = (h16*)alloc((size_t)MP_FP2 * 128 * 2);   // GEMM output: padded
  h16*   xfp1 = (h16*)alloc((size_t)MP_FP1 * 128 * 2);   // GEMM in+out: padded
  int*   ki2  = (int*)alloc((size_t)BATCH * NS1 * 3 * 4);
  float* kw2  = (float*)alloc((size_t)BATCH * NS1 * 3 * 4);
  int*   ki1  = (int*)alloc((size_t)BATCH * NPTS * 3 * 4);
  float* kw1  = (float*)alloc((size_t)BATCH * NPTS * 3 * 4);
  float* meanb= (float*)alloc((size_t)BATCH * 128 * 4);
  h16*   sfa  = (h16*)alloc((size_t)MP_TINY * 64 * 2);   // padded
  h16*   sfeat= (h16*)alloc((size_t)MP_TINY * 64 * 2);   // padded
  h16*   stin = (h16*)alloc((size_t)MP_TINY * 32 * 2);   // padded
  h16*   fin  = (h16*)alloc((size_t)MP_TINY * 192 * 2);  // padded
  float* fout = (float*)alloc((size_t)MP_TINY * 16 * 4); // padded
  for (int i = 0; i < 22; i++) {
    Ls[i].Wp = (h16*)alloc((size_t)Ls[i].Kp * Ls[i].Np * 2);
    Ls[i].bp = (float*)alloc((size_t)Ls[i].Np * 4);
  }
  const size_t PPB = (size_t)10 * 1024 * 1024;  // 10 MiB each (max padded tile ~9.57 MB)
  h16* ping = (h16*)alloc(PPB);
  h16* pong = (h16*)alloc(PPB);

  auto cdiv = [](long a, long b) { return (int)((a + b - 1) / b); };
#define LL(kern, n, ...) kern<<<cdiv((long)(n), 256), 256, 0, stream>>>(__VA_ARGS__)

  // pack all weights/biases to f16 fragment layout
  for (int i = 0; i < 22; i++) {
    int tot = Ls[i].Kp * Ls[i].Np;
    k_pack_w<<<cdiv(tot, 256), 256, 0, stream>>>(Ls[i].W, Ls[i].K, Ls[i].N,
                                                 Ls[i].Kp, Ls[i].Np, Ls[i].Wp);
    k_pack_b<<<cdiv(Ls[i].Np, 256), 256, 0, stream>>>(Ls[i].b, Ls[i].N, Ls[i].Np, Ls[i].bp);
  }

  // block = 128 threads = 4 waves; each block covers 256 rows of one N-tile.
  // M is rounded up to a multiple of 256 inside (buffers are padded).
  auto gemm = [&](const h16* X, L& l, h16* Yh, float* Yf, int M, int relu) {
    dim3 gr(cdiv(M, 256), l.Np / 16);
    k_wmma_gemm<<<gr, 128, 0, stream>>>(X, l.Wp, l.bp, Yh, Yf, l.Kp, l.Np, relu);
  };

  // --- geometry ---
  LL(k_prep, BATCH * NPTS, pc, pos0, feat0);
  k_fps<<<BATCH, 256, 0, stream>>>(pos0, NPTS, NS1, idx1);
  LL(k_gather_pos, BATCH * NS1, pos0, idx1, pos1, NS1, NPTS);
  k_fps<<<BATCH, 256, 0, stream>>>(pos1, NS1, NS2, idx2);
  LL(k_gather_pos, BATCH * NS2, pos1, idx2, pos2, NS2, NS1);
  k_neighbors<<<BATCH * NS1, 128, 0, stream>>>(pos0, pos1, NPTS, NS1, 0.04f, n1i, n1v);
  k_neighbors<<<BATCH * NS2, 128, 0, stream>>>(pos1, pos2, NS1, NS2, 0.16f, n2i, n2v);

  // --- sa1 (per batch, ping/pong; Mb = 37184, padded grid covers 37376) ---
  for (int b = 0; b < BATCH; b++) {
    int Mb = NS1 * KNB;
    LL(k_build_sa_in, (long)Mb * 32, feat0, 3, pos0, pos1, n1i, NS1, NPTS, 32, ping, b);
    gemm(ping, Ls[11], pong, nullptr, Mb, 1);
    gemm(pong, Ls[12], ping, nullptr, Mb, 1);
    gemm(ping, Ls[13], pong, nullptr, Mb, 0);
    LL(k_maxpool_nbr, NS1 * 128, pong, n1v, NS1, 128, x1, b);
  }
  // --- sa2 (per batch; Mb = 9344 -> 9472) ---
  for (int b = 0; b < BATCH; b++) {
    int Mb = NS2 * KNB;
    LL(k_build_sa_in, (long)Mb * 160, x1, 128, pos1, pos2, n2i, NS2, NS1, 160, ping, b);
    gemm(ping, Ls[14], pong, nullptr, Mb, 1);
    gemm(pong, Ls[15], ping, nullptr, Mb, 1);
    gemm(ping, Ls[16], pong, nullptr, Mb, 0);
    LL(k_maxpool_nbr, NS2 * 256, pong, n2v, NS2, 256, x2, b);
  }
  // --- sa3 + global max -> g ---
  {
    int M = BATCH * NS2;
    LL(k_build_sa3in, (long)M * 288, x2, pos2, ping);
    gemm(ping, Ls[17], pong, nullptr, M, 1);
    gemm(pong, Ls[18], ping, nullptr, M, 1);
    gemm(ping, Ls[19], pong, nullptr, M, 0);
    LL(k_maxpool_rows, BATCH * 1024, pong, NS2, 1024, g);
  }
  // --- fp3 (k=1 interp == broadcast g) ---
  {
    int M = BATCH * NS2;
    LL(k_build_fp3in, (long)M * 1280, g, x2, ping);
    gemm(ping, Ls[6], pong, nullptr, M, 1);
    gemm(pong, Ls[7], xfp3, nullptr, M, 0);
  }
  // --- fp2 ---
  LL(k_knn3, BATCH * NS1, pos1, pos2, NS1, NS2, ki2, kw2);
  {
    int M = BATCH * NS1;
    LL(k_build_fp2in, (long)M * 384, xfp3, ki2, kw2, x1, ping);
    gemm(ping, Ls[4], pong, nullptr, M, 1);
    gemm(pong, Ls[5], xfp2, nullptr, M, 0);
  }
  // --- fp1 ---
  LL(k_knn3, BATCH * NPTS, pos0, pos1, NPTS, NS1, ki1, kw1);
  {
    int M = BATCH * NPTS;
    LL(k_build_fp1in, (long)M * 160, xfp2, ki1, kw1, feat0, ping);
    gemm(ping, Ls[1], pong, nullptr, M, 1);
    gemm(pong, Ls[2], ping, nullptr, M, 1);
    gemm(ping, Ls[3], xfp1, nullptr, M, 0);
  }
  // --- head -> flow ---
  {
    int M = BATCH * NPTS;
    gemm(xfp1, Ls[8], ping, nullptr, M, 1);
    gemm(ping, Ls[9], pong, nullptr, M, 1);
    gemm(pong, Ls[10], nullptr, (float*)ping, M, 0);
    LL(k_extract_flow, (long)M * 3, (const float*)ping, (float*)d_out);
  }
  // --- state / finger ---
  LL(k_mean, BATCH * 128, xfp1, meanb);
  LL(k_build_statein, BATCH * 32, eefp, eefq, grip, stin);
  gemm(stin, Ls[20], sfa, nullptr, BATCH, 1);
  gemm(sfa, Ls[21], sfeat, nullptr, BATCH, 0);
  LL(k_build_fingerin, BATCH * 192, meanb, sfeat, fin);
  gemm(fin, Ls[0], nullptr, fout, BATCH, 0);
  k_extract_finger<<<1, 32, 0, stream>>>(fout, (float*)d_out + (size_t)BATCH * NPTS * 3);
#undef LL
}
